// demorgan_64235530879349
// MI455X (gfx1250) — matrix operand order repaired and checked by reading, not compiled, not run on hardware
//
#include <hip/hip_runtime.h>

#define IN_DIM   300
#define MID_DIM  450
#define BATCH    131072
#define KX       320   // padded 300
#define KM       480   // padded 450
#define WSZ      (480 * 320)   // elements per padded weight block (same both shapes)

typedef __attribute__((ext_vector_type(16))) __bf16 bf16x16;
typedef __attribute__((ext_vector_type(8)))  float  f32x8;

__device__ __forceinline__ unsigned short f32_to_bf16_rne(float f) {
    unsigned int u = __float_as_uint(f);
    u += 0x7FFFu + ((u >> 16) & 1u);   // round-to-nearest-even
    return (unsigned short)(u >> 16);
}

union Frag {
    bf16x16 v;
    uint4   q[2];
};

// Warm L2/WGP$ for the next weight panel (global_prefetch_b8).
__device__ __forceinline__ void prefetch_w(const unsigned short* W, int lane) {
    __builtin_prefetch((const void*)(W + (size_t)lane * 128), 0, 1);            // 32 x 256B
    __builtin_prefetch((const void*)(W + (size_t)lane * 128 + 4096), 0, 1);     // next 8KB
}

// ---------------------------------------------------------------------------
// One dense layer applied to TWO 16-row bands: out = leaky(in @ W.T + b).
// Each B-fragment (1KB/wave from global/L2) feeds two wmmas -> 2x weight reuse
// and two independent accumulator chains (no D->C RAW between them).
// in*/out*: LDS bf16 [16][KPAD] / [16][NPAD] row-major.  W: global bf16
// [NPAD][KPAD] row-major zero-padded.
// ---------------------------------------------------------------------------
template<int KPAD, int NPAD>
__device__ __forceinline__ void layer_store2(
    const unsigned short* __restrict__ W,
    const float* __restrict__ bias,
    const unsigned short* in0, const unsigned short* in1,
    unsigned short* out0, unsigned short* out1,
    int lane)
{
    const int nl    = lane & 15;   // N (B/C/D) or M (A) index
    const int kh    = lane >> 4;   // K-half selector
    const int mbase = kh * 8;      // C/D row base
    const unsigned short* arow0 = in0 + nl * KPAD + kh * 16;
    const unsigned short* arow1 = in1 + nl * KPAD + kh * 16;

    for (int n0 = 0; n0 < NPAD; n0 += 16) {
        f32x8 acc0 = {0.f, 0.f, 0.f, 0.f, 0.f, 0.f, 0.f, 0.f};
        f32x8 acc1 = {0.f, 0.f, 0.f, 0.f, 0.f, 0.f, 0.f, 0.f};
        const uint4* wrow = (const uint4*)(W + (size_t)(n0 + nl) * KPAD + kh * 16);
        #pragma unroll
        for (int k0 = 0; k0 < KPAD; k0 += 32) {
            Frag b, a0, a1;
            b.q[0]  = wrow[0];                          // K = k0 + kh*16 .. +7
            b.q[1]  = wrow[1];                          // K = k0 + kh*16+8 .. +15
            wrow   += 4;                                // advance 32 bf16
            a0.q[0] = *(const uint4*)(arow0 + k0);      // K = k0 + kh*16 .. +7
            a0.q[1] = *(const uint4*)(arow0 + k0 + 8);  // K = k0 + kh*16+8 .. +15
            a1.q[0] = *(const uint4*)(arow1 + k0);
            a1.q[1] = *(const uint4*)(arow1 + k0 + 8);
            acc0 = __builtin_amdgcn_wmma_f32_16x16x32_bf16(
                     false, a0.v, false, b.v, (short)0, acc0, false, false);
            acc1 = __builtin_amdgcn_wmma_f32_16x16x32_bf16(
                     false, a1.v, false, b.v, (short)0, acc1, false, false);
        }
        float bv = bias[n0 + nl];
        #pragma unroll
        for (int r = 0; r < 8; ++r) {
            float v0 = acc0[r] + bv;
            v0 = v0 > 0.f ? v0 : 0.1f * v0;             // LeakyReLU(0.1)
            out0[(mbase + r) * NPAD + n0 + nl] = f32_to_bf16_rne(v0);
            float v1 = acc1[r] + bv;
            v1 = v1 > 0.f ? v1 : 0.1f * v1;
            out1[(mbase + r) * NPAD + n0 + nl] = f32_to_bf16_rne(v1);
        }
    }
}

// Final layer of a chain for two bands: fuse squared-error vs x (f32).
template<int KPAD>
__device__ __forceinline__ void layer_loss2(
    const unsigned short* __restrict__ W,     // [320][KPAD] padded
    const float* __restrict__ bias,           // [320] padded
    const unsigned short* in0, const unsigned short* in1,
    const float* __restrict__ x,
    int row0, float& lossacc, int lane)
{
    const int nl    = lane & 15;
    const int kh    = lane >> 4;
    const int mbase = kh * 8;
    const unsigned short* arow0 = in0 + nl * KPAD + kh * 16;
    const unsigned short* arow1 = in1 + nl * KPAD + kh * 16;

    for (int n0 = 0; n0 < 304; n0 += 16) {            // 19 tiles cover n<300
        f32x8 acc0 = {0.f, 0.f, 0.f, 0.f, 0.f, 0.f, 0.f, 0.f};
        f32x8 acc1 = {0.f, 0.f, 0.f, 0.f, 0.f, 0.f, 0.f, 0.f};
        const uint4* wrow = (const uint4*)(W + (size_t)(n0 + nl) * KPAD + kh * 16);
        #pragma unroll
        for (int k0 = 0; k0 < KPAD; k0 += 32) {
            Frag b, a0, a1;
            b.q[0]  = wrow[0];
            b.q[1]  = wrow[1];
            wrow   += 4;
            a0.q[0] = *(const uint4*)(arow0 + k0);
            a0.q[1] = *(const uint4*)(arow0 + k0 + 8);
            a1.q[0] = *(const uint4*)(arow1 + k0);
            a1.q[1] = *(const uint4*)(arow1 + k0 + 8);
            acc0 = __builtin_amdgcn_wmma_f32_16x16x32_bf16(
                     false, a0.v, false, b.v, (short)0, acc0, false, false);
            acc1 = __builtin_amdgcn_wmma_f32_16x16x32_bf16(
                     false, a1.v, false, b.v, (short)0, acc1, false, false);
        }
        int n = n0 + nl;
        if (n < IN_DIM) {
            float bv = bias[n];
            #pragma unroll
            for (int r = 0; r < 8; ++r) {
                float v0 = acc0[r] + bv;
                v0 = v0 > 0.f ? v0 : 0.1f * v0;
                float xv0 = x[(size_t)(row0 + mbase + r) * IN_DIM + n];
                float d0 = xv0 - v0;
                lossacc += d0 * d0;
                float v1 = acc1[r] + bv;
                v1 = v1 > 0.f ? v1 : 0.1f * v1;
                float xv1 = x[(size_t)(row0 + 16 + mbase + r) * IN_DIM + n];
                float d1 = xv1 - v1;
                lossacc += d1 * d1;
            }
        }
    }
}

// ---------------------------------------------------------------------------
// Fused chain kernel: one wave per 32-row (two-band) slice.
// Chain order AND -> OR -> NOT lets bufX double as the NOT chain's pong
// buffer (bufX dead after not1 consumes it): 51.2KB LDS/block -> ~6
// blocks/WGP while halving global weight traffic vs one band/wave.
// ---------------------------------------------------------------------------
__global__ __launch_bounds__(32) void demorgan_fused_kernel(
    const float* __restrict__ x,
    const unsigned short* __restrict__ wb,
    const float* __restrict__ bb,
    float* __restrict__ out)
{
    __shared__ __align__(16) unsigned short bufX[2][16 * KX];  // x / NOT pong
    __shared__ __align__(16) unsigned short buf1[2][16 * KM];  // 480-wide ping

    const int lane = threadIdx.x;
    const int row0 = blockIdx.x * 32;

    const unsigned short* Wn1 = wb + 0 * WSZ;
    const unsigned short* Wn2 = wb + 1 * WSZ;
    const unsigned short* Wa1 = wb + 2 * WSZ;
    const unsigned short* Wa2 = wb + 3 * WSZ;
    const unsigned short* Wo1 = wb + 4 * WSZ;
    const unsigned short* Wo2 = wb + 5 * WSZ;
    const float* bn1 = bb + 0;    const float* bn2 = bb + 480;
    const float* ba1 = bb + 800;  const float* ba2 = bb + 1280;
    const float* bo1 = bb + 1600; const float* bo2 = bb + 2080;

    prefetch_w(Wa1, lane);

    // Stage both x bands -> bf16 LDS with zero K-padding (uniform trips: 320).
    for (int idx = lane; idx < 2 * 16 * KX; idx += 32) {
        int band = idx / (16 * KX);
        int rem  = idx - band * (16 * KX);
        int m = rem / KX, k = rem - m * KX;
        float v = (k < IN_DIM) ? x[(size_t)(row0 + band * 16 + m) * IN_DIM + k] : 0.f;
        bufX[band][rem] = f32_to_bf16_rne(v);
    }
    // Single wave owns its LDS buffers: per-wave DS ordering + compiler waits.

    float loss = 0.f;
    // AND([x,x]) with folded W1
    layer_store2<KX, KM>(Wa1, ba1, bufX[0], bufX[1], buf1[0], buf1[1], lane);
    prefetch_w(Wo1, lane);
    layer_loss2 <KM>    (Wa2, ba2, buf1[0], buf1[1], x, row0, loss, lane);
    // OR([x,x]) with folded W1
    layer_store2<KX, KM>(Wo1, bo1, bufX[0], bufX[1], buf1[0], buf1[1], lane);
    prefetch_w(Wn1, lane);
    layer_loss2 <KM>    (Wo2, bo2, buf1[0], buf1[1], x, row0, loss, lane);
    // NOT(NOT(x)) chain; bufX is dead after not1 -> reuse as 320-wide pong
    layer_store2<KX, KM>(Wn1, bn1, bufX[0], bufX[1], buf1[0], buf1[1], lane);
    layer_store2<KM, KX>(Wn2, bn2, buf1[0], buf1[1], bufX[0], bufX[1], lane);
    layer_store2<KX, KM>(Wn1, bn1, bufX[0], bufX[1], buf1[0], buf1[1], lane);
    layer_loss2 <KM>    (Wn2, bn2, buf1[0], buf1[1], x, row0, loss, lane);

    // wave32 reduction, one atomic per 32-row slice
    loss += __shfl_xor(loss, 16, 32);
    loss += __shfl_xor(loss, 8, 32);
    loss += __shfl_xor(loss, 4, 32);
    loss += __shfl_xor(loss, 2, 32);
    loss += __shfl_xor(loss, 1, 32);
    if (lane == 0) atomicAdd(out, loss);
}

// ---------------------------------------------------------------------------
// Prep: f32 weights -> padded bf16 (folding [x,x] first layers), pad biases.
// ---------------------------------------------------------------------------
__global__ void prep_weight_kernel(const float* __restrict__ src,
                                   unsigned short* __restrict__ dst,
                                   int O, int K, int Opad, int Kpad,
                                   int srcld, int foldOff)
{
    int i = blockIdx.x * blockDim.x + threadIdx.x;
    if (i >= Opad * Kpad) return;
    int o = i / Kpad, k = i - o * Kpad;
    float v = 0.f;
    if (o < O && k < K) {
        v = src[(size_t)o * srcld + k];
        if (foldOff) v += src[(size_t)o * srcld + k + foldOff];
    }
    dst[i] = f32_to_bf16_rne(v);
}

__global__ void prep_bias_kernel(const float* __restrict__ src,
                                 float* __restrict__ dst, int O, int Opad)
{
    int i = blockIdx.x * blockDim.x + threadIdx.x;
    if (i < Opad) dst[i] = (i < O) ? src[i] : 0.f;
}

__global__ void zero_out_kernel(float* out) {
    if (threadIdx.x == 0 && blockIdx.x == 0) out[0] = 0.f;
}

__global__ void finalize_kernel(float* out) {
    if (threadIdx.x == 0 && blockIdx.x == 0)
        out[0] *= (float)(1.0 / (3.0 * (double)BATCH * (double)IN_DIM));
}

// ---------------------------------------------------------------------------
extern "C" void kernel_launch(void* const* d_in, const int* in_sizes, int n_in,
                              void* d_out, int out_size, void* d_ws, size_t ws_size,
                              hipStream_t stream)
{
    (void)in_sizes; (void)n_in; (void)out_size; (void)ws_size;
    const float* x   = (const float*)d_in[0];
    const float* nW1 = (const float*)d_in[1];
    const float* nb1 = (const float*)d_in[2];
    const float* nW2 = (const float*)d_in[3];
    const float* nb2 = (const float*)d_in[4];
    const float* aW1 = (const float*)d_in[5];
    const float* ab1 = (const float*)d_in[6];
    const float* aW2 = (const float*)d_in[7];
    const float* ab2 = (const float*)d_in[8];
    const float* oW1 = (const float*)d_in[9];
    const float* ob1 = (const float*)d_in[10];
    const float* oW2 = (const float*)d_in[11];
    const float* ob2 = (const float*)d_in[12];
    float* out = (float*)d_out;

    unsigned short* wbp = (unsigned short*)d_ws;                      // 6*WSZ bf16
    float* bbp = (float*)((char*)d_ws + (size_t)6 * WSZ * sizeof(unsigned short));

    dim3 pblk(256), pgrd((WSZ + 255) / 256);
    prep_weight_kernel<<<pgrd, pblk, 0, stream>>>(nW1, wbp + 0 * WSZ, 450, 300, 480, 320, 300, 0);
    prep_weight_kernel<<<pgrd, pblk, 0, stream>>>(nW2, wbp + 1 * WSZ, 300, 450, 320, 480, 450, 0);
    prep_weight_kernel<<<pgrd, pblk, 0, stream>>>(aW1, wbp + 2 * WSZ, 450, 300, 480, 320, 600, 300);
    prep_weight_kernel<<<pgrd, pblk, 0, stream>>>(aW2, wbp + 3 * WSZ, 300, 450, 320, 480, 450, 0);
    prep_weight_kernel<<<pgrd, pblk, 0, stream>>>(oW1, wbp + 4 * WSZ, 450, 300, 480, 320, 600, 300);
    prep_weight_kernel<<<pgrd, pblk, 0, stream>>>(oW2, wbp + 5 * WSZ, 300, 450, 320, 480, 450, 0);
    prep_bias_kernel<<<2, 256, 0, stream>>>(nb1, bbp + 0,    450, 480);
    prep_bias_kernel<<<2, 256, 0, stream>>>(nb2, bbp + 480,  300, 320);
    prep_bias_kernel<<<2, 256, 0, stream>>>(ab1, bbp + 800,  450, 480);
    prep_bias_kernel<<<2, 256, 0, stream>>>(ab2, bbp + 1280, 300, 320);
    prep_bias_kernel<<<2, 256, 0, stream>>>(ob1, bbp + 1600, 450, 480);
    prep_bias_kernel<<<2, 256, 0, stream>>>(ob2, bbp + 2080, 300, 320);

    zero_out_kernel<<<1, 1, 0, stream>>>(out);
    demorgan_fused_kernel<<<BATCH / 32, 32, 0, stream>>>(x, wbp, bbp, out);
    finalize_kernel<<<1, 1, 0, stream>>>(out);
}
